// KOSLMCell_47528108098137
// MI455X (gfx1250) — matrix operand order repaired and checked by reading, not compiled, not run on hardware
//
#include <hip/hip_runtime.h>

// ---------------------------------------------------------------------------
// KOSLM cell for MI455X (gfx1250): bf16 WMMA GEMMs, TDM (tensor_load_to_lds)
// double-buffered tile staging, fused epilogues. B=16384, D=1024, H=2048.
// ---------------------------------------------------------------------------

typedef unsigned short ushort_t;
typedef __attribute__((ext_vector_type(16))) __bf16 v16bf;
typedef __attribute__((ext_vector_type(8)))  float  v8f;
typedef __attribute__((ext_vector_type(4)))  unsigned int v4u;
typedef __attribute__((ext_vector_type(8)))  int v8i;
typedef __attribute__((ext_vector_type(4)))  int v4i;

#define B_SZ 16384
#define D_SZ 1024
#define H_SZ 2048

#ifndef __has_builtin
#define __has_builtin(x) 0
#endif
#if __has_builtin(__builtin_amdgcn_tensor_load_to_lds) && \
    __has_builtin(__builtin_amdgcn_s_wait_tensorcnt)
#define USE_TDM 1
#else
#define USE_TDM 0
#endif

// ---- fp32 -> bf16 (round-to-nearest-even) ---------------------------------
__device__ __forceinline__ ushort_t f2bf(float f) {
  unsigned u = __float_as_uint(f);
  u += 0x7FFFu + ((u >> 16) & 1u);
  return (ushort_t)(u >> 16);
}

// ---------------------------------------------------------------------------
// Packing / conversion kernels
// ---------------------------------------------------------------------------
__global__ __launch_bounds__(256) void k_pack_act(const float* __restrict__ x,
                                                  const float* __restrict__ h,
                                                  ushort_t* __restrict__ dst) {
  unsigned idx = blockIdx.x * 256u + threadIdx.x;       // < B*(D+H)
  unsigned b = idx / 3072u, c = idx % 3072u;
  float v = (c < D_SZ) ? x[(size_t)b * D_SZ + c]
                       : h[(size_t)b * H_SZ + (c - D_SZ)];
  dst[idx] = f2bf(v);
}

__global__ __launch_bounds__(256) void k_pack_w(const float* __restrict__ Wih,
                                                const float* __restrict__ Whh,
                                                ushort_t* __restrict__ dst) {
  unsigned idx = blockIdx.x * 256u + threadIdx.x;       // < 4096*3072
  unsigned n = idx / 3072u, c = idx % 3072u;
  float v = (c < D_SZ) ? Wih[(size_t)n * D_SZ + c]
                       : Whh[(size_t)n * H_SZ + (c - D_SZ)];
  dst[idx] = f2bf(v);
}

__global__ __launch_bounds__(256) void k_cvt_bf16(const float* __restrict__ src,
                                                  ushort_t* __restrict__ dst,
                                                  unsigned n) {
  unsigned i = blockIdx.x * 256u + threadIdx.x;
  if (i < n) dst[i] = f2bf(src[i]);
}

__global__ __launch_bounds__(256) void k_bias_cat(const float* __restrict__ bih,
                                                  const float* __restrict__ bhh,
                                                  float* __restrict__ dst) {
  unsigned i = blockIdx.x * 256u + threadIdx.x;
  if (i < 4096u) dst[i] = bih[i] + bhh[i];
}

// ---------------------------------------------------------------------------
// Pointwise: z_t, M_t, innov from raw gates
// ---------------------------------------------------------------------------
__global__ __launch_bounds__(256) void k_pointwise(const float* __restrict__ gates,
                                                   const float* __restrict__ c_prev,
                                                   const float* __restrict__ z_bias,
                                                   const float* __restrict__ m_bias,
                                                   const float* __restrict__ loglam,
                                                   float* __restrict__ zout,
                                                   float* __restrict__ Mout,
                                                   ushort_t* __restrict__ innov) {
  unsigned i = blockIdx.x * 256u + threadIdx.x;         // < B*H
  unsigned b = i >> 11, h = i & 2047u;
  float zr = gates[(size_t)b * 4096u + h];
  float mr = gates[(size_t)b * 4096u + 2048u + h];
  float z  = tanhf(zr + z_bias[h]);
  float M  = 1.5f * tanhf(mr + m_bias[h]);
  float Ab = -expf(loglam[h]);
  float inn = z - M * (Ab * c_prev[i]);
  inn = fminf(fmaxf(inn, -20.0f), 20.0f);
  zout[i] = z;
  Mout[i] = M;
  innov[i] = f2bf(inn);
}

// ---------------------------------------------------------------------------
// TDM descriptor helpers (CDNA5 D#, CDNA5 ISA)
// 2-D tile: 128 rows x 64B (8 x 8-byte elements), row stride K bf16 elems.
// LDS padding: pad_enable, interval 64B (pad_interval=3), amount 16B
// (pad_amount=3 -> 4 DWORDs)  => LDS row stride 80B == LDSP*2.
// ---------------------------------------------------------------------------
#if USE_TDM
__device__ __forceinline__ v4u tdm_g0(unsigned lds_addr, unsigned long long ga) {
  v4u g;
  g.x = 1u;                                  // count=1, user mode, no gather
  g.y = lds_addr;                            // LDS byte address
  g.z = (unsigned)ga;                        // global_addr[31:0]
  g.w = (unsigned)((ga >> 32) & 0x01FFFFFFu) // global_addr[56:32]
        | (2u << 30);                        // type = 2 ("image")
  return g;
}

__device__ __forceinline__ v8i tdm_g1(int K) {
  v8i g;
  g[0] = (int)((3u << 16)      // data_size = 8 bytes
             | (1u << 20)      // pad_enable
             | (3u << 22)      // pad_interval: every 16 DWORDs (64B)
             | (3u << 25));    // pad_amount: 4 DWORDs (16B)
  g[1] = (int)0xFFFF0000u;     // tensor_dim0 = 0xFFFF (low16 @ [31:16])
  g[2] = (int)0xFFFF0000u;     // tensor_dim1 = 0xFFFF (low16 @ [31:16])
  g[3] = (int)(8u << 16);      // tile_dim0 = 8 (8B elems per row)
  g[4] = 128;                  // tile_dim1 = 128 rows, tile_dim2 = 0
  g[5] = K / 4;                // tensor_dim0_stride low32 (8B units) = K*2/8
  g[6] = 0;                    // stride0 hi16 = 0, stride1 low16 = 0
  g[7] = 0;                    // stride1 hi32 = 0
  return g;
}
#endif

// ---------------------------------------------------------------------------
// WMMA GEMM: C[M,N] = A[M,K] @ W[N,K]^T (+ bias), fused epilogue by MODE.
//   MODE 0: store fp32 (raw gates)
//   MODE 1: exact gelu -> bf16 (hid)
//   MODE 2: full cell update -> c_pre fp32 (written into d_out c-slot)
// Block tile 128x128x32; 8 wave32s, each wave a 64x32 region (4x2 WMMA tiles).
// Tiles staged by the Tensor Data Mover, double-buffered on TENSORcnt.
// ---------------------------------------------------------------------------
template <int MODE>
__global__ __launch_bounds__(256)
void k_gemm(const ushort_t* __restrict__ A, const ushort_t* __restrict__ W,
            const float* __restrict__ bias, int K, int N,
            float* __restrict__ outF, ushort_t* __restrict__ outBf,
            const float* __restrict__ c_prev, const float* __restrict__ zbuf,
            const float* __restrict__ Mbuf, const float* __restrict__ loglam) {
  constexpr int BM = 128, BN = 128, BK = 32, LDSP = 40;  // +8 elem pad: 16B-aligned, conflict-free
  __shared__ ushort_t sA[2][BM * LDSP];
  __shared__ ushort_t sB[2][BN * LDSP];

  const int tid  = threadIdx.x;
  const int lane = tid & 31;
  const int wave = tid >> 5;
  const int wr   = wave & 1;    // 64-row slab
  const int wc   = wave >> 1;   // 32-col slab
  const int half = lane >> 4;
  const int lrow = lane & 15;
  const size_t rowBase = (size_t)blockIdx.y * BM;
  const size_t colBase = (size_t)blockIdx.x * BN;
  const int niter = K / BK;

  v8f zero;
  #pragma unroll
  for (int i = 0; i < 8; ++i) zero[i] = 0.0f;
  v8f acc[4][2];
  #pragma unroll
  for (int mt = 0; mt < 4; ++mt)
    #pragma unroll
    for (int nt = 0; nt < 2; ++nt) acc[mt][nt] = zero;

  union FU { v16bf v; uint4 q[2]; };

#if USE_TDM
  // generic->LDS: low 32 bits of the generic address are the LDS byte offset
  const unsigned ldsA[2] = { (unsigned)(size_t)&sA[0][0], (unsigned)(size_t)&sA[1][0] };
  const unsigned ldsB[2] = { (unsigned)(size_t)&sB[0][0], (unsigned)(size_t)&sB[1][0] };
  const unsigned long long gaA = (unsigned long long)(size_t)(A + rowBase * (size_t)K);
  const unsigned long long gaB = (unsigned long long)(size_t)(W + colBase * (size_t)K);
  const v8i g1 = tdm_g1(K);
  const v4i gz4 = {0, 0, 0, 0};
  const v8i gz8 = {0, 0, 0, 0, 0, 0, 0, 0};

  if (wave == 0) {  // TDM ignores EXEC; issue once per block from wave 0
    __builtin_amdgcn_tensor_load_to_lds(tdm_g0(ldsA[0], gaA), g1, gz4, gz4, gz8, 0);
    __builtin_amdgcn_tensor_load_to_lds(tdm_g0(ldsB[0], gaB), g1, gz4, gz4, gz8, 0);
  }

  for (int it = 0; it < niter; ++it) {
    const int cur = it & 1;
    if (wave == 0) {
      if (it + 1 < niter) {  // prefetch next tile pair into the other buffer
        unsigned long long koff = (unsigned long long)(it + 1) * (BK * 2);
        __builtin_amdgcn_tensor_load_to_lds(tdm_g0(ldsA[cur ^ 1], gaA + koff), g1, gz4, gz4, gz8, 0);
        __builtin_amdgcn_tensor_load_to_lds(tdm_g0(ldsB[cur ^ 1], gaB + koff), g1, gz4, gz4, gz8, 0);
        __builtin_amdgcn_s_wait_tensorcnt(2);  // current pair done, next in flight
      } else {
        __builtin_amdgcn_s_wait_tensorcnt(0);
      }
    }
    __syncthreads();  // publish LDS tile to all waves

    const ushort_t* sAb = sA[cur];
    const ushort_t* sBb = sB[cur];
#else
  for (int it = 0; it < niter; ++it) {
    const int k0 = it * BK;
    __syncthreads();
    #pragma unroll
    for (int i = 0; i < 2; ++i) {
      int c = tid + i * 256;
      int row = c >> 2, part = c & 3;
      const ushort_t* ga = A + (rowBase + (size_t)row) * K + k0 + part * 8;
      const ushort_t* gb = W + (colBase + (size_t)row) * K + k0 + part * 8;
      *reinterpret_cast<uint4*>(&sA[0][row * LDSP + part * 8]) =
          *reinterpret_cast<const uint4*>(ga);
      *reinterpret_cast<uint4*>(&sB[0][row * LDSP + part * 8]) =
          *reinterpret_cast<const uint4*>(gb);
    }
    __syncthreads();
    const ushort_t* sAb = sA[0];
    const ushort_t* sBb = sB[0];
#endif

    // --- fragments per ISA bf16 lane layouts -------------------------------
    FU fa[4], fb[2];
    #pragma unroll
    for (int mt = 0; mt < 4; ++mt) {
      int base = (wr * 64 + mt * 16 + lrow) * LDSP;
      fa[mt].q[0] = *reinterpret_cast<const uint4*>(&sAb[base + half * 8]);       // K=half*8..+7
      fa[mt].q[1] = *reinterpret_cast<const uint4*>(&sAb[base + 16 + half * 8]);  // K=16+half*8..+7
    }
    #pragma unroll
    for (int nt = 0; nt < 2; ++nt) {
      int base = (wc * 32 + nt * 16 + lrow) * LDSP + half * 8;                    // K=half*8..+7, 16+half*8..+7
      fb[nt].q[0] = *reinterpret_cast<const uint4*>(&sBb[base]);
      fb[nt].q[1] = *reinterpret_cast<const uint4*>(&sBb[base + 16]);
    }

    #pragma unroll
    for (int mt = 0; mt < 4; ++mt)
      #pragma unroll
      for (int nt = 0; nt < 2; ++nt)
        acc[mt][nt] = __builtin_amdgcn_wmma_f32_16x16x32_bf16(
            false, fa[mt].v, false, fb[nt].v, (short)0, acc[mt][nt], false, false);

    __syncthreads();  // all waves done reading before buffer reuse
  }

  // --- epilogue: D layout: VGPR r -> M = r + 8*half, N = lane&15 -----------
  #pragma unroll
  for (int mt = 0; mt < 4; ++mt) {
    #pragma unroll
    for (int nt = 0; nt < 2; ++nt) {
      size_t gn = colBase + (size_t)(wc * 32 + nt * 16 + lrow);
      float bn = bias[gn];
      #pragma unroll
      for (int r = 0; r < 8; ++r) {
        size_t gm = rowBase + (size_t)(wr * 64 + mt * 16 + r + half * 8);
        float v = acc[mt][nt][r] + bn;
        if (MODE == 0) {
          outF[gm * (size_t)N + gn] = v;
        } else if (MODE == 1) {
          float g = 0.5f * v * (1.0f + erff(v * 0.70710678118654752f));
          outBf[gm * (size_t)N + gn] = f2bf(g);
        } else {
          size_t idx = gm * (size_t)N + gn;  // N == H
          float Kt = tanhf(v) * 0.45f;       // k_scale * 0.5
          float Mv = fminf(fmaxf(Mbuf[idx], -1.5f), 1.5f);
          float Ab = -expf(loglam[gn]);
          float At = fminf(fmaxf((1.0f - Kt * Mv) * Ab, 0.0f), 0.95f);
          float Bt = fminf(fmaxf(Kt, -0.95f), 0.95f);
          outF[idx] = At * c_prev[idx] + Bt * zbuf[idx];
        }
      }
    }
  }
}

// ---------------------------------------------------------------------------
// Row RMS clamp + h_t = M_t * c_t  (one block per row, in-place on d_out)
// ---------------------------------------------------------------------------
__global__ __launch_bounds__(256) void k_norm(const float* __restrict__ Mbuf,
                                              float* __restrict__ out) {
  __shared__ float red[256];
  const int row = blockIdx.x;
  const int tid = threadIdx.x;
  float* cptr = out + (size_t)B_SZ * H_SZ + (size_t)row * H_SZ;
  float* hptr = out + (size_t)row * H_SZ;
  float vals[8];
  float s = 0.0f;
  #pragma unroll
  for (int i = 0; i < 8; ++i) {
    vals[i] = cptr[tid + i * 256];
    s += vals[i] * vals[i];
  }
  red[tid] = s;
  __syncthreads();
  for (int off = 128; off > 0; off >>= 1) {
    if (tid < off) red[tid] += red[tid + off];
    __syncthreads();
  }
  float mean  = red[0] * (1.0f / H_SZ);
  float norm  = sqrtf(mean + 1e-6f);
  float scale = fminf(100.0f / norm, 1.0f);
  #pragma unroll
  for (int i = 0; i < 8; ++i) {
    float c = vals[i] * scale;
    float Mv = fminf(fmaxf(Mbuf[(size_t)row * H_SZ + tid + i * 256], -1.5f), 1.5f);
    cptr[tid + i * 256] = c;
    hptr[tid + i * 256] = Mv * c;
  }
}

// ---------------------------------------------------------------------------
// Host launch
// ---------------------------------------------------------------------------
extern "C" void kernel_launch(void* const* d_in, const int* in_sizes, int n_in,
                              void* d_out, int out_size, void* d_ws, size_t ws_size,
                              hipStream_t stream) {
  const float* x        = (const float*)d_in[0];
  const float* h_prev   = (const float*)d_in[1];
  const float* c_prev   = (const float*)d_in[2];
  const float* W_ih     = (const float*)d_in[3];
  const float* b_ih     = (const float*)d_in[4];
  const float* W_hh     = (const float*)d_in[5];
  const float* b_hh     = (const float*)d_in[6];
  const float* loglam   = (const float*)d_in[7];
  const float* z_bias   = (const float*)d_in[8];
  const float* m_bias   = (const float*)d_in[9];
  const float* W1       = (const float*)d_in[10];
  const float* b1       = (const float*)d_in[11];
  const float* W2       = (const float*)d_in[12];
  const float* b2       = (const float*)d_in[13];
  float* out = (float*)d_out;
  char*  ws  = (char*)d_ws;

  // workspace layout (bytes)
  const size_t OFF_ACAT  = 0;                                       // [B,3072] bf16
  const size_t OFF_WCAT  = OFF_ACAT + (size_t)B_SZ * 3072 * 2;      // [4096,3072] bf16
  const size_t OFF_W1    = OFF_WCAT + (size_t)4096 * 3072 * 2;      // [6144,2048] bf16
  const size_t OFF_W2    = OFF_W1   + (size_t)6144 * 2048 * 2;      // [2048,6144] bf16
  const size_t OFF_BIAS  = OFF_W2   + (size_t)2048 * 6144 * 2;      // [4096] f32
  const size_t OFF_Z     = OFF_BIAS + (size_t)4096 * 4;             // [B,H] f32
  const size_t OFF_M     = OFF_Z    + (size_t)B_SZ * H_SZ * 4;      // [B,H] f32
  const size_t OFF_INNOV = OFF_M    + (size_t)B_SZ * H_SZ * 4;      // [B,H] bf16
  const size_t OFF_BIG   = OFF_INNOV + (size_t)B_SZ * H_SZ * 2;     // gates f32 / hid bf16 (aliased)

  ushort_t* Acat  = (ushort_t*)(ws + OFF_ACAT);
  ushort_t* Wcat  = (ushort_t*)(ws + OFF_WCAT);
  ushort_t* W1b   = (ushort_t*)(ws + OFF_W1);
  ushort_t* W2b   = (ushort_t*)(ws + OFF_W2);
  float*    biasC = (float*)   (ws + OFF_BIAS);
  float*    zbuf  = (float*)   (ws + OFF_Z);
  float*    Mbuf  = (float*)   (ws + OFF_M);
  ushort_t* innov = (ushort_t*)(ws + OFF_INNOV);
  float*    gates = (float*)   (ws + OFF_BIG);
  ushort_t* hid   = (ushort_t*)(ws + OFF_BIG);   // reuses gates region after consumption
  float*    c_slot = out + (size_t)B_SZ * H_SZ;  // c_t half of d_out (in-place norm)

  // 1) pack/convert to bf16
  k_pack_act<<<(B_SZ * 3072u) / 256u, 256, 0, stream>>>(x, h_prev, Acat);
  k_pack_w  <<<(4096u * 3072u) / 256u, 256, 0, stream>>>(W_ih, W_hh, Wcat);
  k_cvt_bf16<<<(6144u * 2048u) / 256u, 256, 0, stream>>>(W1, W1b, 6144u * 2048u);
  k_cvt_bf16<<<(2048u * 6144u) / 256u, 256, 0, stream>>>(W2, W2b, 2048u * 6144u);
  k_bias_cat<<<16, 256, 0, stream>>>(b_ih, b_hh, biasC);

  // 2) gates = [x|h] @ [W_ih|W_hh]^T + (b_ih+b_hh)   (K=3072, N=4096)
  k_gemm<0><<<dim3(4096 / 128, B_SZ / 128), 256, 0, stream>>>(
      Acat, Wcat, biasC, 3072, 4096, gates, nullptr,
      nullptr, nullptr, nullptr, nullptr);

  // 3) pointwise: z_t, M_t, innov
  k_pointwise<<<(B_SZ * H_SZ) / 256u, 256, 0, stream>>>(
      gates, c_prev, z_bias, m_bias, loglam, zbuf, Mbuf, innov);

  // 4) hid = gelu(innov @ W1^T + b1)   (K=2048, N=6144) -> bf16 (aliases gates)
  k_gemm<1><<<dim3(6144 / 128, B_SZ / 128), 256, 0, stream>>>(
      innov, W1b, b1, 2048, 6144, nullptr, hid,
      nullptr, nullptr, nullptr, nullptr);

  // 5) raw_K = hid @ W2^T + b2; fused cell update -> c_pre into d_out c-slot
  k_gemm<2><<<dim3(2048 / 128, B_SZ / 128), 256, 0, stream>>>(
      hid, W2b, b2, 6144, 2048, c_slot, nullptr,
      c_prev, zbuf, Mbuf, loglam);

  // 6) row RMS clamp + h_t = M_t * c_t
  k_norm<<<B_SZ, 256, 0, stream>>>(Mbuf, out);
}
